// BiaffineSpanHead_32779190403866
// MI455X (gfx1250) — matrix operand order repaired and checked by reading, not compiled, not run on hardware
//
#include <hip/hip_runtime.h>

typedef float v2f __attribute__((ext_vector_type(2)));
typedef float v8f __attribute__((ext_vector_type(8)));
typedef __bf16 v16bf __attribute__((ext_vector_type(16)));

#define B_ 8
#define S_ 512
#define D_ 768
#define H_ 256
#define C_ 32

// ---- workspace byte offsets (total 25 MB) --------------------------------
#define HSF_OFF 0u          // Hs  f32  (4096 x 256)  4 MB
#define HEF_OFF 4194304u    // He  f32  (4096 x 256)  4 MB
#define LSF_OFF 8388608u    // linS f32 (4096 x 32)   0.5 MB
#define LEF_OFF 8912896u    // linE f32 (4096 x 32)   0.5 MB
#define HSH_OFF 9437184u    // Hs hi bf16             2 MB
#define HSL_OFF 11534336u   // Hs lo bf16             2 MB
#define HEH_OFF 13631488u   // He hi bf16             2 MB
#define HEL_OFF 15728640u   // He lo bf16             2 MB
#define UTH_OFF 17825792u   // Ut hi bf16 [c][g][h]   4 MB
#define UTL_OFF 22020096u   // Ut lo bf16 [c][g][h]   4 MB

struct Q2 { uint4 a, b; };  // 32B container for one v16bf operand

__device__ __forceinline__ v8f wmma_f32(v2f a, v2f b, v8f c) {
  return __builtin_amdgcn_wmma_f32_16x16x4_f32(false, a, false, b, (short)0, c,
                                               false, false);
}
__device__ __forceinline__ v8f wmma_bf16(v16bf a, v16bf b, v8f c) {
  return __builtin_amdgcn_wmma_f32_16x16x32_bf16(false, a, false, b, (short)0,
                                                 c, false, false);
}
__device__ __forceinline__ unsigned short f2bf(float x) {
  return __builtin_bit_cast(unsigned short, (__bf16)x);
}

// A operand (16-bit, 16x32): lane lo-half needs K = k..k+7 and k+16..k+23,
// hi-half the same +8.  Two 16B chunks from a K-contiguous row.
__device__ __forceinline__ v16bf load_a16(const unsigned short* rowBase, int k,
                                          int hiHalf) {
  Q2 q;
  q.a = *(const uint4*)(rowBase + k + hiHalf * 8);
  q.b = *(const uint4*)(rowBase + k + 16 + hiHalf * 8);
  return __builtin_bit_cast(v16bf, q);
}
// B operand (16-bit, 32x16): lane needs K = p..p+7 and p+16..p+23 (two 16B chunks).
__device__ __forceinline__ v16bf load_b16(const unsigned short* p) {
  Q2 q;
  q.a = *(const uint4*)(p);
  q.b = *(const uint4*)(p + 16);
  return __builtin_bit_cast(v16bf, q);
}

// ---------------------------------------------------------------------------
// Kernel A: Hs/He = X @ W^T + b  (exact f32 WMMA, K=768), emits f32 + bf16
// hi/lo copies of the result for the split-precision biaffine stage.
// ---------------------------------------------------------------------------
__global__ __launch_bounds__(256) void bsh_proj_kernel(
    const float* __restrict__ X, const float* __restrict__ Wsw,
    const float* __restrict__ Wsb, const float* __restrict__ Wew,
    const float* __restrict__ Web, char* __restrict__ ws) {
  const int tid = threadIdx.x;
  const int wave = tid >> 5, lane = tid & 31;
  const int lo = lane & 15, hi = lane >> 4;

  int tile = blockIdx.x * 8 + wave;  // 0..8191
  const int proj = tile >> 12;       // 0 = Hs, 1 = He
  tile &= 4095;
  const int m0 = (tile >> 4) * 16, n0 = (tile & 15) * 16;

  const float* W = proj ? Wew : Wsw;
  const float* bias = proj ? Web : Wsb;
  float* outF = (float*)(ws + (proj ? HEF_OFF : HSF_OFF));
  unsigned short* outH = (unsigned short*)(ws + (proj ? HEH_OFF : HSH_OFF));
  unsigned short* outL = (unsigned short*)(ws + (proj ? HEL_OFF : HSL_OFF));

  const float* aRow = X + (size_t)(m0 + lo) * D_ + hi * 2;
  const float* bRow = W + (size_t)(n0 + lo) * D_ + hi * 2;

  v8f acc = {0.f, 0.f, 0.f, 0.f, 0.f, 0.f, 0.f, 0.f};
#pragma unroll 4
  for (int k = 0; k < D_; k += 4) {
    v2f a = *(const v2f*)(aRow + k);
    v2f b = *(const v2f*)(bRow + k);
    acc = wmma_f32(a, b, acc);
  }
  const float bv = bias[n0 + lo];
#pragma unroll
  for (int v = 0; v < 8; ++v) {
    const size_t o = (size_t)(m0 + v + 8 * hi) * H_ + n0 + lo;
    const float x = acc[v] + bv;
    outF[o] = x;
    const __bf16 hb = (__bf16)x;
    outH[o] = __builtin_bit_cast(unsigned short, hb);
    outL[o] = f2bf(x - (float)hb);
  }
}

// ---------------------------------------------------------------------------
// Kernel B: linS = Hs @ W_s^T + W_b ; linE = He @ W_e^T  (f32 WMMA, tiny)
// ---------------------------------------------------------------------------
__global__ __launch_bounds__(256) void bsh_lin_kernel(
    const float* __restrict__ Ww, const float* __restrict__ Wb, char* ws) {
  const int tid = threadIdx.x;
  const int wave = tid >> 5, lane = tid & 31;
  const int lo = lane & 15, hi = lane >> 4;

  int tile = blockIdx.x * 8 + wave;  // 0..1023
  const int proj = tile >> 9;
  tile &= 511;
  const int m0 = (tile >> 1) * 16, n0 = (tile & 1) * 16;

  const float* Hsrc = (const float*)(ws + (proj ? HEF_OFF : HSF_OFF));
  float* out = (float*)(ws + (proj ? LEF_OFF : LSF_OFF));

  const float* aRow = Hsrc + (size_t)(m0 + lo) * H_ + hi * 2;
  const float* bRow = Ww + (size_t)(n0 + lo) * (2 * H_) + proj * H_ + hi * 2;

  v8f acc = {0.f, 0.f, 0.f, 0.f, 0.f, 0.f, 0.f, 0.f};
#pragma unroll 4
  for (int k = 0; k < H_; k += 4) {
    v2f a = *(const v2f*)(aRow + k);
    v2f b = *(const v2f*)(bRow + k);
    acc = wmma_f32(a, b, acc);
  }
  const float bv = proj ? 0.f : Wb[n0 + lo];  // fold W_b into linS
#pragma unroll
  for (int v = 0; v < 8; ++v)
    out[(size_t)(m0 + v + 8 * hi) * C_ + n0 + lo] = acc[v] + bv;
}

// ---------------------------------------------------------------------------
// Kernel U: transpose U(h,c,g) -> Ut[c][g][h] as bf16 hi/lo (LDS-tiled).
// Makes the step-1 B operand K-contiguous for 32B WMMA operand loads.
// ---------------------------------------------------------------------------
__global__ __launch_bounds__(256) void bsh_ucvt_kernel(
    const float* __restrict__ U, char* __restrict__ ws) {
  __shared__ float t[16 * 260];
  unsigned short* uth = (unsigned short*)(ws + UTH_OFF);
  unsigned short* utl = (unsigned short*)(ws + UTL_OFF);
  const int tid = threadIdx.x;
  const int c = blockIdx.y;        // 0..31
  const int g0 = blockIdx.x * 16;  // 0..240

  for (int idx = tid; idx < 4096; idx += 256) {
    const int g = idx & 15, h = idx >> 4;
    t[g * 260 + h] = U[(size_t)h * (C_ * H_) + c * H_ + g0 + g];
  }
  __syncthreads();
  for (int idx = tid; idx < 4096; idx += 256) {
    const int h = idx & 255, g = idx >> 8;
    const float x = t[g * 260 + h];
    const __bf16 hb = (__bf16)x;
    const size_t o = ((size_t)c * H_ + g0 + g) * H_ + h;
    uth[o] = __builtin_bit_cast(unsigned short, hb);
    utl[o] = f2bf(x - (float)hb);
  }
}

// ---------------------------------------------------------------------------
// Kernel C: out[b,s,e,c] = sum_g (sum_h Hs U) He + linS + linE
// bf16 split-precision (hi*hi + lo*hi + hi*lo), f32 accumulation.
// Grid: (sTile=32, cOctet=4, b=8), 256 threads = 8 waves.
// ---------------------------------------------------------------------------
#define GP 264  // padded LDS row stride (ushorts)

__global__ __launch_bounds__(256) void bsh_biaff_kernel(
    const char* __restrict__ wsb, float* __restrict__ out) {
  __shared__ unsigned short ldsHsHi[16 * GP];   // 8.25 KB
  __shared__ unsigned short ldsHsLo[16 * GP];   // 8.25 KB
  __shared__ unsigned short ldsTHi[128 * GP];   // 66 KB : T[c8][s16][g256] hi
  __shared__ unsigned short ldsTLo[128 * GP];   // 66 KB : lo
  __shared__ float ldsO[16 * 16 * 8];           // 8 KB  : out stage [s][e][c8]

  const int tid = threadIdx.x;
  const int wave = tid >> 5, lane = tid & 31;
  const int lo = lane & 15, hi = lane >> 4;

  const int s0 = blockIdx.x * 16;  // s tile
  const int c0 = blockIdx.y * 8;   // c octet
  const int bb = blockIdx.z;       // batch
  const int row0 = bb * S_ + s0;   // flat row in (4096, *)

  const unsigned short* HsHi = (const unsigned short*)(wsb + HSH_OFF);
  const unsigned short* HsLo = (const unsigned short*)(wsb + HSL_OFF);
  const unsigned short* HeHi = (const unsigned short*)(wsb + HEH_OFF);
  const unsigned short* HeLo = (const unsigned short*)(wsb + HEL_OFF);
  const unsigned short* UtHi = (const unsigned short*)(wsb + UTH_OFF);
  const unsigned short* UtLo = (const unsigned short*)(wsb + UTL_OFF);
  const float* LS = (const float*)(wsb + LSF_OFF);
  const float* LE = (const float*)(wsb + LEF_OFF);

  // --- stage Hs hi/lo tile (16 x 256 bf16 each) into LDS ---
  for (int idx = tid; idx < 16 * 32; idx += 256) {
    const int r = idx >> 5, c8 = (idx & 31) * 8;
    *(uint4*)&ldsHsHi[r * GP + c8] =
        *(const uint4*)(HsHi + (size_t)(row0 + r) * H_ + c8);
    *(uint4*)&ldsHsLo[r * GP + c8] =
        *(const uint4*)(HsLo + (size_t)(row0 + r) * H_ + c8);
  }
  __syncthreads();

  // --- step 1: T[c][s][g] = Hs_tile x U[:, c0+c, :]  (8 c x 16 g-tiles) ---
  for (int t = wave; t < 128; t += 8) {
    const int c = t >> 4;
    const int g0 = (t & 15) * 16;
    const unsigned short* bh = UtHi + ((size_t)(c0 + c) * H_ + g0 + lo) * H_;
    const unsigned short* bl = UtLo + ((size_t)(c0 + c) * H_ + g0 + lo) * H_;
    const unsigned short* aHiRow = &ldsHsHi[lo * GP];
    const unsigned short* aLoRow = &ldsHsLo[lo * GP];
    v8f acc = {0.f, 0.f, 0.f, 0.f, 0.f, 0.f, 0.f, 0.f};
#pragma unroll 2
    for (int k = 0; k < H_; k += 32) {
      const v16bf ah = load_a16(aHiRow, k, hi);
      const v16bf al = load_a16(aLoRow, k, hi);
      const v16bf bhi = load_b16(bh + k + hi * 8);
      const v16bf blo = load_b16(bl + k + hi * 8);
      acc = wmma_bf16(ah, bhi, acc);
      acc = wmma_bf16(al, bhi, acc);
      acc = wmma_bf16(ah, blo, acc);
    }
    unsigned short* tHi = &ldsTHi[(c * 16) * GP + g0 + lo];
    unsigned short* tLo = &ldsTLo[(c * 16) * GP + g0 + lo];
#pragma unroll
    for (int v = 0; v < 8; ++v) {
      const float x = acc[v];
      const __bf16 hb = (__bf16)x;
      tHi[(v + 8 * hi) * GP] = __builtin_bit_cast(unsigned short, hb);
      tLo[(v + 8 * hi) * GP] = f2bf(x - (float)hb);
    }
  }
  __syncthreads();

  // --- step 2: per e-tile, out_tile = T[c] x He_tile^T + lin ---
  const int c = c0 + wave;  // one c channel per wave
  const unsigned short* aHiRow = &ldsTHi[(wave * 16 + lo) * GP];
  const unsigned short* aLoRow = &ldsTLo[(wave * 16 + lo) * GP];
  for (int eT = 0; eT < 32; ++eT) {
    const int e0 = eT * 16;
    const unsigned short* bh = HeHi + (size_t)(bb * S_ + e0 + lo) * H_;
    const unsigned short* bl = HeLo + (size_t)(bb * S_ + e0 + lo) * H_;
    if (eT < 31) {  // prefetch next e-tile (global_prefetch_b8)
      __builtin_prefetch(HeHi + (size_t)(bb * S_ + e0 + 16 + lo) * H_, 0, 1);
      __builtin_prefetch(HeLo + (size_t)(bb * S_ + e0 + 16 + lo) * H_, 0, 1);
    }
    v8f acc = {0.f, 0.f, 0.f, 0.f, 0.f, 0.f, 0.f, 0.f};
#pragma unroll 2
    for (int k = 0; k < H_; k += 32) {
      const v16bf ah = load_a16(aHiRow, k, hi);
      const v16bf al = load_a16(aLoRow, k, hi);
      const v16bf bhi = load_b16(bh + k + hi * 8);
      const v16bf blo = load_b16(bl + k + hi * 8);
      acc = wmma_bf16(ah, bhi, acc);
      acc = wmma_bf16(al, bhi, acc);
      acc = wmma_bf16(ah, blo, acc);
    }
    const float le = LE[(size_t)(bb * S_ + e0 + lo) * C_ + c];
#pragma unroll
    for (int v = 0; v < 8; ++v) {
      const int m = v + 8 * hi;
      const float ls = LS[(size_t)(row0 + m) * C_ + c];
      ldsO[(m * 16 + lo) * 8 + wave] = acc[v] + ls + le;
    }
    __syncthreads();

    // cooperative store: one (s,e) pair per thread -> 32B-contiguous c-run
    {
      const int m = tid >> 4, n = tid & 15;
      const float4 p0 = *(const float4*)(&ldsO[(m * 16 + n) * 8]);
      const float4 p1 = *(const float4*)(&ldsO[(m * 16 + n) * 8 + 4]);
      float* dst =
          out + ((size_t)(bb * S_ + s0 + m) * S_ + (e0 + n)) * C_ + c0;
      *(float4*)dst = p0;
      *(float4*)(dst + 4) = p1;
    }
    __syncthreads();
  }
}

// ---------------------------------------------------------------------------
extern "C" void kernel_launch(void* const* d_in, const int* in_sizes, int n_in,
                              void* d_out, int out_size, void* d_ws,
                              size_t ws_size, hipStream_t stream) {
  const float* X   = (const float*)d_in[0];  // seq_feats (8,512,768)
  const float* U   = (const float*)d_in[1];  // (256,32,256)
  const float* Ww  = (const float*)d_in[2];  // (32,512)
  const float* Wb  = (const float*)d_in[3];  // (32)
  const float* Wsw = (const float*)d_in[4];  // (256,768)
  const float* Wsb = (const float*)d_in[5];  // (256)
  const float* Wew = (const float*)d_in[6];  // (256,768)
  const float* Web = (const float*)d_in[7];  // (256)
  float* out = (float*)d_out;                // (8,512,512,32)
  char* ws = (char*)d_ws;                    // needs >= 25 MB

  bsh_proj_kernel<<<1024, 256, 0, stream>>>(X, Wsw, Wsb, Wew, Web, ws);
  bsh_lin_kernel<<<128, 256, 0, stream>>>(Ww, Wb, ws);
  bsh_ucvt_kernel<<<dim3(16, 32), 256, 0, stream>>>(U, ws);
  bsh_biaff_kernel<<<dim3(32, 4, 8), 256, 0, stream>>>(ws, out);
}
